// ImportanceGuidedAttention_22651657519406
// MI455X (gfx1250) — matrix operand order repaired and checked by reading, not compiled, not run on hardware
//
#include <hip/hip_runtime.h>
#include <hip/hip_bf16.h>
#include <stdint.h>

typedef __attribute__((ext_vector_type(16))) _Float16 v16h;
typedef __attribute__((ext_vector_type(8)))  float    v8f;

union AF { v16h v; uint4 q[2]; };

constexpr int Bn  = 4;
constexpr int Sn  = 2048;
constexpr int HDn = 1024;
constexpr int Hn  = 16;
constexpr int Dn  = 64;
constexpr int Mn  = Bn * Sn;      // 8192

// ---------------- CDNA5 async global->LDS path (probe via __has_builtin) ----------------
#if __has_builtin(__builtin_amdgcn_global_load_async_to_lds_b128)
#define USE_ASYNC 1
typedef int ai4 __attribute__((vector_size(16)));
static __device__ __forceinline__ void async_copy_b128(const _Float16* g, _Float16* l) {
    __builtin_amdgcn_global_load_async_to_lds_b128(
        (__attribute__((address_space(1))) ai4*)(__attribute__((address_space(1))) void*)(void*)const_cast<_Float16*>(g),
        (__attribute__((address_space(3))) ai4*)(__attribute__((address_space(3))) void*)(void*)l,
        0, 0);
}
#else
#define USE_ASYNC 0
#endif

static __device__ __forceinline__ void wait_asynccnt0() {
#if __has_builtin(__builtin_amdgcn_s_wait_asynccnt)
    __builtin_amdgcn_s_wait_asynccnt(0);
#else
    asm volatile("s_wait_asynccnt 0" ::: "memory");
#endif
}

// ---------------- workspace layout (bytes) ----------------
constexpr size_t OFF_XH    = 0;                                  // X  f16 [M,HD]
constexpr size_t OFF_WQH   = OFF_XH   + (size_t)Mn*HDn*2;        // W  f16 [HD,HD] x4
constexpr size_t OFF_WKH   = OFF_WQH  + (size_t)HDn*HDn*2;
constexpr size_t OFF_WVH   = OFF_WKH  + (size_t)HDn*HDn*2;
constexpr size_t OFF_WOH   = OFF_WVH  + (size_t)HDn*HDn*2;
constexpr size_t OFF_QH    = OFF_WOH  + (size_t)HDn*HDn*2;       // Q  f16 [B,S,HD]
constexpr size_t OFF_KH    = OFF_QH   + (size_t)Mn*HDn*2;        // K  f16 [B,S,HD]
constexpr size_t OFF_KF    = OFF_KH   + (size_t)Mn*HDn*2;        // K  f32 [B,S,HD]
constexpr size_t OFF_VF    = OFF_KF   + (size_t)Mn*HDn*4;        // V  f32 [B,S,HD]
constexpr size_t OFF_ATTNH = OFF_VF   + (size_t)Mn*HDn*4;        // attn f16 [B,S,HD]
constexpr size_t OFF_CKC   = OFF_ATTNH+ (size_t)Mn*HDn*2;        // ck compact f16 [B,H,S,D]
constexpr size_t OFF_CVCT  = OFF_CKC  + (size_t)Mn*HDn*2;        // cv compact f16 [B,H,D,S]
constexpr size_t OFF_IMP   = OFF_CVCT + (size_t)Mn*HDn*2;        // imp f32 [B,S]
constexpr size_t OFF_LAB   = OFF_IMP  + (size_t)Bn*Sn*4;         // labels i32 [B,S]
constexpr size_t OFF_SEL   = OFF_LAB  + (size_t)Bn*Sn*4;         // sel i32 [B,S]
constexpr size_t OFF_BIAS  = OFF_SEL  + (size_t)Bn*Sn*4;         // bias f32 [B]
constexpr size_t OFF_SCAL  = OFF_BIAS + 256;                     // ksel, n32
constexpr size_t OFF_QP    = OFF_SCAL + 256;                     // 12 floats quant params
constexpr size_t OFF_PART  = OFF_QP   + 256;                     // 256*12 f32 partial min/max

// ---------------- f32 -> f16 conversion ----------------
__global__ __launch_bounds__(256) void f32_to_f16_kernel(const float* __restrict__ in,
                                                         _Float16* __restrict__ out, size_t n) {
    size_t i = (size_t)blockIdx.x * 256 + threadIdx.x;
    size_t stride = (size_t)gridDim.x * 256;
    for (; i < n; i += stride) out[i] = (_Float16)in[i];
}

// ---------------- WMMA GEMM: C = A[f16 MxK] * B[f16 KxN] ----------------
// Double-buffered LDS; A tile staged via async global->LDS (ASYNCcnt) when available,
// B tile staged via registers with transpose-scatter into Bs[n][k].
__global__ __launch_bounds__(256) void gemm_wmma_kernel(
    const _Float16* __restrict__ A, const _Float16* __restrict__ Bm,
    float* __restrict__ Cf, _Float16* __restrict__ Ch, int M, int N, int K) {
    __shared__ __align__(16) _Float16 As[2][128][40];   // [buf][row][k], padded
    __shared__ __align__(16) _Float16 Bs[2][128][40];   // [buf][n][k] (transposed), padded
    const int tid  = threadIdx.x;
    const int lane = tid & 31;
    const int w    = tid >> 5;
    const int hi   = (lane >> 4) & 1;
    const int ln   = lane & 15;
    const int m_blk = blockIdx.x * 128;
    const int n_blk = blockIdx.y * 128;
    const int wm0 = (w & 3) * 32;        // wave row group (32 rows)
    const int wn0 = (w >> 2) * 64;       // wave col group (64 cols)
    const int ca = tid & 3,  ra  = tid >> 2;   // A staging coords (16B chunk, row)
    const int cb = tid & 15, kb0 = tid >> 4;   // B staging coords (8-half chunk along N, k row)

    v8f acc[2][4] = {};
    uint4 breg[2];
#if !USE_ASYNC
    uint4 areg[2];
#endif

    // ---- prologue: stage tile 0 into buffer 0
#if USE_ASYNC
    async_copy_b128(A + (size_t)(m_blk + ra) * K + ca * 8,      &As[0][ra][ca * 8]);
    async_copy_b128(A + (size_t)(m_blk + ra + 64) * K + ca * 8, &As[0][ra + 64][ca * 8]);
#else
    areg[0] = *(const uint4*)(A + (size_t)(m_blk + ra) * K + ca * 8);
    areg[1] = *(const uint4*)(A + (size_t)(m_blk + ra + 64) * K + ca * 8);
    *(uint4*)&As[0][ra][ca * 8]      = areg[0];
    *(uint4*)&As[0][ra + 64][ca * 8] = areg[1];
#endif
    breg[0] = *(const uint4*)(Bm + (size_t)(kb0) * N + n_blk + cb * 8);
    breg[1] = *(const uint4*)(Bm + (size_t)(kb0 + 16) * N + n_blk + cb * 8);
    {
        const _Float16* h0 = (const _Float16*)&breg[0];
        const _Float16* h1 = (const _Float16*)&breg[1];
        #pragma unroll
        for (int i = 0; i < 8; i++) { Bs[0][cb * 8 + i][kb0] = h0[i]; Bs[0][cb * 8 + i][kb0 + 16] = h1[i]; }
    }
#if USE_ASYNC
    wait_asynccnt0();
#endif
    __syncthreads();

    for (int kk = 0; kk < K; kk += 32) {
        const int cur = (kk >> 5) & 1, nxt = cur ^ 1;
        const bool has_next = (kk + 32) < K;
        if (has_next) {   // issue next-tile loads (overlap with WMMA below)
#if USE_ASYNC
            async_copy_b128(A + (size_t)(m_blk + ra) * K + kk + 32 + ca * 8,      &As[nxt][ra][ca * 8]);
            async_copy_b128(A + (size_t)(m_blk + ra + 64) * K + kk + 32 + ca * 8, &As[nxt][ra + 64][ca * 8]);
#else
            areg[0] = *(const uint4*)(A + (size_t)(m_blk + ra) * K + kk + 32 + ca * 8);
            areg[1] = *(const uint4*)(A + (size_t)(m_blk + ra + 64) * K + kk + 32 + ca * 8);
#endif
            breg[0] = *(const uint4*)(Bm + (size_t)(kk + 32 + kb0) * N + n_blk + cb * 8);
            breg[1] = *(const uint4*)(Bm + (size_t)(kk + 32 + kb0 + 16) * N + n_blk + cb * 8);
            if (kk + 64 < K)   // keep L2 a tile ahead -> global_prefetch_b8
                __builtin_prefetch(Bm + (size_t)(kk + 64 + kb0) * N + n_blk + cb * 8, 0, 1);
        }
        // fragments from current buffer
        AF afr[2], bfr[4];
        #pragma unroll
        for (int mi = 0; mi < 2; mi++) {
            int row  = wm0 + mi * 16 + ln;
            int off0 = hi ? 8 : 0;                       // A layout: lo lanes K0-7/16-23, hi lanes K8-15/24-31
            afr[mi].q[0] = *(const uint4*)&As[cur][row][off0];
            afr[mi].q[1] = *(const uint4*)&As[cur][row][off0 + 16];
        }
        #pragma unroll
        for (int ni = 0; ni < 4; ni++) {
            int col  = wn0 + ni * 16 + ln;
            int koff = hi ? 8 : 0;                       // B layout: lo lanes K0-7/16-23, hi lanes K8-15/24-31
            bfr[ni].q[0] = *(const uint4*)&Bs[cur][col][koff];
            bfr[ni].q[1] = *(const uint4*)&Bs[cur][col][koff + 16];
        }
        #pragma unroll
        for (int mi = 0; mi < 2; mi++)
            #pragma unroll
            for (int ni = 0; ni < 4; ni++)
                acc[mi][ni] = __builtin_amdgcn_wmma_f32_16x16x32_f16(
                    false, afr[mi].v, false, bfr[ni].v, (short)0, acc[mi][ni], false, false);
        if (has_next) {   // commit next tile into the other buffer
#if !USE_ASYNC
            *(uint4*)&As[nxt][ra][ca * 8]      = areg[0];
            *(uint4*)&As[nxt][ra + 64][ca * 8] = areg[1];
#endif
            const _Float16* h0 = (const _Float16*)&breg[0];
            const _Float16* h1 = (const _Float16*)&breg[1];
            #pragma unroll
            for (int i = 0; i < 8; i++) { Bs[nxt][cb * 8 + i][kb0] = h0[i]; Bs[nxt][cb * 8 + i][kb0 + 16] = h1[i]; }
#if USE_ASYNC
            wait_asynccnt0();
#endif
        }
        __syncthreads();
    }
    #pragma unroll
    for (int mi = 0; mi < 2; mi++)
        #pragma unroll
        for (int ni = 0; ni < 4; ni++)
            #pragma unroll
            for (int r = 0; r < 8; r++) {
                int row = m_blk + wm0 + mi * 16 + r + (hi ? 8 : 0);
                int col = n_blk + wn0 + ni * 16 + ln;
                float v = acc[mi][ni][r];
                if (Cf) Cf[(size_t)row * N + col] = v;
                if (Ch) Ch[(size_t)row * N + col] = (_Float16)v;
            }
}

// ---------------- per-batch bias: mean_s(q0) . w_imp + b_imp ----------------
__global__ __launch_bounds__(64) void bias_kernel(const _Float16* __restrict__ Qh,
                                                  const float* __restrict__ w_imp,
                                                  const float* __restrict__ b_imp,
                                                  float* __restrict__ bias) {
    __shared__ float red[64];
    int b = blockIdx.x, tid = threadIdx.x;
    float s = 0.f;
    for (int srow = 0; srow < Sn; srow++)
        s += (float)Qh[((size_t)(b * Sn + srow)) * HDn + tid];   // head 0, dim tid
    red[tid] = (s / (float)Sn) * w_imp[tid];
    __syncthreads();
    if (tid == 0) {
        float t = 0.f;
        for (int i = 0; i < 64; i++) t += red[i];
        bias[b] = t + b_imp[0];
    }
}

// ---------------- importance: flash-style masked softmax over head-0 scores ----------------
__global__ __launch_bounds__(256) void importance_kernel(
    const _Float16* __restrict__ Qh, const _Float16* __restrict__ Kh,
    const unsigned char* __restrict__ pm, const float* __restrict__ bias,
    float* __restrict__ imp, int* __restrict__ labels) {
    const int tid = threadIdx.x, lane = tid & 31, w = tid >> 5;
    const int hi = (lane >> 4) & 1, ln = lane & 15;
    const int b = blockIdx.x;
    const int qbase = blockIdx.y * 128 + w * 16;

    AF aq[2];
    #pragma unroll
    for (int kb = 0; kb < 2; kb++) {
        const _Float16* p = Qh + ((size_t)(b * Sn + qbase + ln)) * HDn;   // head 0
        int d0 = kb * 32 + (hi ? 8 : 0);
        aq[kb].q[0] = *(const uint4*)(p + d0);
        aq[kb].q[1] = *(const uint4*)(p + d0 + 16);
    }
    float m[8], num[8], den[8];
    #pragma unroll
    for (int r = 0; r < 8; r++) { m[r] = -1e30f; num[r] = 0.f; den[r] = 0.f; }
    const float scale = 0.125f;   // 1/sqrt(64)

    for (int t0 = 0; t0 < Sn; t0 += 16) {
        AF bk[2];
        #pragma unroll
        for (int kb = 0; kb < 2; kb++) {
            const _Float16* p = Kh + ((size_t)(b * Sn + t0 + ln)) * HDn;  // head 0
            int d0 = kb * 32 + (hi ? 8 : 0);
            bk[kb].q[0] = *(const uint4*)(p + d0);
            bk[kb].q[1] = *(const uint4*)(p + d0 + 16);
        }
        v8f c = {};
        c = __builtin_amdgcn_wmma_f32_16x16x32_f16(false, aq[0].v, false, bk[0].v, (short)0, c, false, false);
        c = __builtin_amdgcn_wmma_f32_16x16x32_f16(false, aq[1].v, false, bk[1].v, (short)0, c, false, false);
        float pmv = (float)pm[t0 + ln];
        #pragma unroll
        for (int r = 0; r < 8; r++) {
            float z = c[r] * scale;
            float tmax = z;
            for (int off = 1; off < 16; off <<= 1) tmax = fmaxf(tmax, __shfl_xor(tmax, off, 32));
            float nm = fmaxf(m[r], tmax);
            float e  = __expf(z - nm);
            float ep = e * pmv;
            for (int off = 1; off < 16; off <<= 1) { e += __shfl_xor(e, off, 32); ep += __shfl_xor(ep, off, 32); }
            float corr = __expf(m[r] - nm);
            den[r] = den[r] * corr + e;
            num[r] = num[r] * corr + ep;
            m[r] = nm;
        }
    }
    if (ln == 0) {
        float bs = bias[b];
        #pragma unroll
        for (int r = 0; r < 8; r++) {
            int row = qbase + r + (hi ? 8 : 0);
            float v = num[r] / den[r] + bs;
            imp[b * Sn + row] = v;
            labels[b * Sn + row] = (v > 0.7f) ? 2 : ((v >= 0.3f) ? 1 : 0);
        }
    }
}

// ---------------- median (q=0.5 linear interp) + k_sel ----------------
__global__ __launch_bounds__(1024) void median_kernel(const float* __restrict__ imp,
                                                      int* __restrict__ ksel, int* __restrict__ n32) {
    __shared__ float arr[Bn * Sn];
    __shared__ int cnt[Bn];
    __shared__ float thr_s;
    int tid = threadIdx.x;
    for (int i = tid; i < Bn * Sn; i += 1024) arr[i] = imp[i];
    __syncthreads();
    for (int k = 2; k <= Bn * Sn; k <<= 1)
        for (int j = k >> 1; j > 0; j >>= 1) {
            for (int i = tid; i < Bn * Sn; i += 1024) {
                int ixj = i ^ j;
                if (ixj > i) {
                    bool up = ((i & k) == 0);
                    float a = arr[i], c = arr[ixj];
                    if ((a > c) == up) { arr[i] = c; arr[ixj] = a; }
                }
            }
            __syncthreads();
        }
    if (tid == 0) {
        thr_s = 0.5f * (arr[Bn * Sn / 2 - 1] + arr[Bn * Sn / 2]);
        for (int i = 0; i < Bn; i++) cnt[i] = 0;
    }
    __syncthreads();
    float thr = thr_s;
    for (int i = tid; i < Bn * Sn; i += 1024)
        if (imp[i] >= thr) atomicAdd(&cnt[i / Sn], 1);
    __syncthreads();
    if (tid == 0) {
        int ks = cnt[0];
        for (int i = 1; i < Bn; i++) ks = (cnt[i] > ks) ? cnt[i] : ks;
        if (ks < 1) ks = 1;
        if (ks > Sn) ks = Sn;
        *ksel = ks;
        *n32  = (ks + 31) / 32;
    }
}

// ---------------- per-batch descending sort of (imp, idx) -> sel ----------------
__global__ __launch_bounds__(1024) void topk_kernel(const float* __restrict__ imp,
                                                    int* __restrict__ sel) {
    __shared__ float v[Sn];
    __shared__ int   ix[Sn];
    int b = blockIdx.x, tid = threadIdx.x;
    for (int i = tid; i < Sn; i += 1024) { v[i] = imp[b * Sn + i]; ix[i] = i; }
    __syncthreads();
    for (int k = 2; k <= Sn; k <<= 1)
        for (int j = k >> 1; j > 0; j >>= 1) {
            for (int i = tid; i < Sn; i += 1024) {
                int ixj = i ^ j;
                if (ixj > i) {
                    bool desc = ((i & k) == 0);
                    float a = v[i], c = v[ixj];
                    int ai = ix[i], ci = ix[ixj];
                    bool a_first = (a > c) || (a == c && ai < ci);  // descending order, stable
                    if (desc ? !a_first : a_first) { v[i] = c; v[ixj] = a; ix[i] = ci; ix[ixj] = ai; }
                }
            }
            __syncthreads();
        }
    for (int i = tid; i < Sn; i += 1024) sel[b * Sn + i] = ix[i];
}

// ---------------- per-level masked min/max, stage 1 ----------------
__global__ __launch_bounds__(256) void minmax1_kernel(const float* __restrict__ Kf,
                                                      const float* __restrict__ Vf,
                                                      const int* __restrict__ labels,
                                                      float* __restrict__ part) {
    float mn[6], mx[6];
    #pragma unroll
    for (int i = 0; i < 6; i++) { mn[i] = 1e30f; mx[i] = -1e30f; }
    size_t total = (size_t)Bn * Sn * HDn;
    for (size_t idx = (size_t)blockIdx.x * 256 + threadIdx.x; idx < total; idx += (size_t)gridDim.x * 256) {
        int lab = labels[idx >> 10];   // HDn = 1024
        float kv = Kf[idx], vv = Vf[idx];
        mn[lab]     = fminf(mn[lab], kv);     mx[lab]     = fmaxf(mx[lab], kv);
        mn[3 + lab] = fminf(mn[3 + lab], vv); mx[3 + lab] = fmaxf(mx[3 + lab], vv);
    }
    __shared__ float smn[6][256], smx[6][256];
    int tid = threadIdx.x;
    for (int i = 0; i < 6; i++) { smn[i][tid] = mn[i]; smx[i][tid] = mx[i]; }
    __syncthreads();
    for (int s = 128; s > 0; s >>= 1) {
        if (tid < s)
            for (int i = 0; i < 6; i++) {
                smn[i][tid] = fminf(smn[i][tid], smn[i][tid + s]);
                smx[i][tid] = fmaxf(smx[i][tid], smx[i][tid + s]);
            }
        __syncthreads();
    }
    if (tid == 0)
        for (int i = 0; i < 6; i++) {
            part[blockIdx.x * 12 + i]     = smn[i][0];
            part[blockIdx.x * 12 + 6 + i] = smx[i][0];
        }
}

// ---------------- min/max stage 2 -> quant params ----------------
__global__ __launch_bounds__(256) void minmax2_kernel(const float* __restrict__ part,
                                                      float* __restrict__ qp) {
    __shared__ float smn[6][256], smx[6][256];
    int tid = threadIdx.x;
    for (int i = 0; i < 6; i++) { smn[i][tid] = part[tid * 12 + i]; smx[i][tid] = part[tid * 12 + 6 + i]; }
    __syncthreads();
    for (int s = 128; s > 0; s >>= 1) {
        if (tid < s)
            for (int i = 0; i < 6; i++) {
                smn[i][tid] = fminf(smn[i][tid], smn[i][tid + s]);
                smx[i][tid] = fmaxf(smx[i][tid], smx[i][tid + s]);
            }
        __syncthreads();
    }
    if (tid == 0) {
        const float qmaxs[3] = {3.f, 15.f, 255.f};
        for (int t = 0; t < 2; t++)
            for (int l = 0; l < 3; l++) {
                float tmin = smn[t * 3 + l][0], tmax = smx[t * 3 + l][0];
                bool degen = (tmin > 0.9e30f) || (tmax < -0.9e30f) || (tmax == tmin);
                float sc = degen ? 1.f : (tmax - tmin) / qmaxs[l];
                float zp = degen ? 0.f : -tmin / sc;
                qp[t * 6 + l]     = sc;
                qp[t * 6 + 3 + l] = zp;
            }
    }
}

// ---------------- quant-dequant + gather into compact ck [B,H,S,D] / cv^T [B,H,D,S] ----------------
__global__ __launch_bounds__(256) void gather_quant_kernel(
    const float* __restrict__ Kf, const float* __restrict__ Vf,
    const int* __restrict__ labels, const int* __restrict__ sel,
    const float* __restrict__ qp, const int* __restrict__ kselp,
    _Float16* __restrict__ ckc, _Float16* __restrict__ cvct) {
    int j = blockIdx.x, b = blockIdx.y;
    int ks = *kselp;
    if (j >= ks) return;
    int s   = sel[b * Sn + j];
    int lab = labels[b * Sn + s];
    float qmax = (lab == 0) ? 3.f : ((lab == 1) ? 15.f : 255.f);
    float ksc = qp[lab], kzp = qp[3 + lab], vsc = qp[6 + lab], vzp = qp[9 + lab];
    for (int hd = threadIdx.x; hd < HDn; hd += 256) {
        int h = hd >> 6, d = hd & 63;
        float kv = Kf[((size_t)(b * Sn + s)) * HDn + hd];
        float dqk = (fminf(fmaxf(rintf(kv / ksc + kzp), 0.f), qmax) - kzp) * ksc;
        ckc[(((size_t)(b * Hn + h)) * Sn + j) * Dn + d] = (_Float16)dqk;
        float vv = Vf[((size_t)(b * Sn + s)) * HDn + hd];
        float dqv = (fminf(fmaxf(rintf(vv / vsc + vzp), 0.f), qmax) - vzp) * vsc;
        cvct[(((size_t)(b * Hn + h)) * Dn + d) * Sn + j] = (_Float16)dqv;
    }
}

// ---------------- flash attention over selected quantized KV ----------------
__global__ __launch_bounds__(256) void attention_kernel(
    const _Float16* __restrict__ Qh, const _Float16* __restrict__ ckc,
    const _Float16* __restrict__ cvct, const int* __restrict__ kselp,
    const int* __restrict__ n32p, _Float16* __restrict__ attn) {
    __shared__ __align__(16) _Float16 Plds[8][16][32];   // per-wave P tile
    const int tid = threadIdx.x, lane = tid & 31, w = tid >> 5;
    const int hi = (lane >> 4) & 1, ln = lane & 15;
    const int bh = blockIdx.y;
    const int b = bh >> 4, h = bh & 15;
    const int qbase = blockIdx.x * 128 + w * 16;
    const int ks = *kselp, n32 = *n32p;

    AF aq[2];
    #pragma unroll
    for (int kb = 0; kb < 2; kb++) {
        const _Float16* p = Qh + ((size_t)(b * Sn + qbase + ln)) * HDn + h * Dn;
        int d0 = kb * 32 + (hi ? 8 : 0);
        aq[kb].q[0] = *(const uint4*)(p + d0);
        aq[kb].q[1] = *(const uint4*)(p + d0 + 16);
    }
    v8f o[4] = {};
    float m[8], l[8];
    #pragma unroll
    for (int r = 0; r < 8; r++) { m[r] = -1e30f; l[r] = 0.f; }
    const float scale = 0.125f;
    const size_t ckbase = (size_t)bh * Sn * Dn;
    const size_t cvbase = (size_t)bh * Dn * Sn;

    for (int t = 0; t < n32; t++) {
        int t0 = t * 32;
        v8f sc[2];
        #pragma unroll
        for (int kg = 0; kg < 2; kg++) {
            AF bk[2];
            #pragma unroll
            for (int kb = 0; kb < 2; kb++) {
                const _Float16* p = ckc + ckbase + (size_t)(t0 + kg * 16 + ln) * Dn;
                int d0 = kb * 32 + (hi ? 8 : 0);
                bk[kb].q[0] = *(const uint4*)(p + d0);
                bk[kb].q[1] = *(const uint4*)(p + d0 + 16);
            }
            v8f c = {};
            c = __builtin_amdgcn_wmma_f32_16x16x32_f16(false, aq[0].v, false, bk[0].v, (short)0, c, false, false);
            c = __builtin_amdgcn_wmma_f32_16x16x32_f16(false, aq[1].v, false, bk[1].v, (short)0, c, false, false);
            sc[kg] = c;
        }
        bool v0 = (t0 + ln) < ks, v1 = (t0 + 16 + ln) < ks;
        #pragma unroll
        for (int r = 0; r < 8; r++) {
            float z0 = v0 ? sc[0][r] * scale : -1e30f;
            float z1 = v1 ? sc[1][r] * scale : -1e30f;
            float tmax = fmaxf(z0, z1);
            for (int off = 1; off < 16; off <<= 1) tmax = fmaxf(tmax, __shfl_xor(tmax, off, 32));
            float nm = fmaxf(m[r], tmax);
            float p0 = __expf(z0 - nm), p1 = __expf(z1 - nm);
            float ss = p0 + p1;
            for (int off = 1; off < 16; off <<= 1) ss += __shfl_xor(ss, off, 32);
            float corr = __expf(m[r] - nm);
            l[r] = l[r] * corr + ss;
            m[r] = nm;
            #pragma unroll
            for (int nb = 0; nb < 4; nb++) o[nb][r] *= corr;
            int row = r + (hi ? 8 : 0);
            Plds[w][row][ln]      = (_Float16)p0;
            Plds[w][row][16 + ln] = (_Float16)p1;
        }
        asm volatile("s_wait_dscnt 0" ::: "memory");   // wave-private LDS: P stores -> A-frag loads
        AF pf;
        {
            int off0 = hi ? 8 : 0;
            pf.q[0] = *(const uint4*)&Plds[w][ln][off0];
            pf.q[1] = *(const uint4*)&Plds[w][ln][off0 + 16];
        }
        #pragma unroll
        for (int nb = 0; nb < 4; nb++) {
            AF vf;
            const _Float16* p = cvct + cvbase + (size_t)(nb * 16 + ln) * Sn + t0 + (hi ? 8 : 0);
            vf.q[0] = *(const uint4*)(p);
            vf.q[1] = *(const uint4*)(p + 16);
            o[nb] = __builtin_amdgcn_wmma_f32_16x16x32_f16(false, pf.v, false, vf.v, (short)0, o[nb], false, false);
        }
    }
    #pragma unroll
    for (int nb = 0; nb < 4; nb++)
        #pragma unroll
        for (int r = 0; r < 8; r++) {
            int row = qbase + r + (hi ? 8 : 0);
            float v = o[nb][r] / l[r];
            attn[((size_t)(b * Sn + row)) * HDn + h * Dn + nb * 16 + ln] = (_Float16)v;
        }
}

// ---------------- host-side orchestration ----------------
extern "C" void kernel_launch(void* const* d_in, const int* in_sizes, int n_in,
                              void* d_out, int out_size, void* d_ws, size_t ws_size,
                              hipStream_t stream) {
    (void)in_sizes; (void)n_in; (void)out_size; (void)ws_size;
    const float* X    = (const float*)d_in[0];
    const float* Wq   = (const float*)d_in[1];
    const float* Wk   = (const float*)d_in[2];
    const float* Wv   = (const float*)d_in[3];
    const float* Wo   = (const float*)d_in[4];
    const float* wimp = (const float*)d_in[5];
    const float* bimp = (const float*)d_in[6];
    const unsigned char* pm = (const unsigned char*)d_in[7];
    float* out = (float*)d_out;
    char* ws = (char*)d_ws;

    _Float16* Xh   = (_Float16*)(ws + OFF_XH);
    _Float16* Wqh  = (_Float16*)(ws + OFF_WQH);
    _Float16* Wkh  = (_Float16*)(ws + OFF_WKH);
    _Float16* Wvh  = (_Float16*)(ws + OFF_WVH);
    _Float16* Woh  = (_Float16*)(ws + OFF_WOH);
    _Float16* Qh   = (_Float16*)(ws + OFF_QH);
    _Float16* Kh   = (_Float16*)(ws + OFF_KH);
    float*    Kf   = (float*)(ws + OFF_KF);
    float*    Vf   = (float*)(ws + OFF_VF);
    _Float16* Ah   = (_Float16*)(ws + OFF_ATTNH);
    _Float16* ckc  = (_Float16*)(ws + OFF_CKC);
    _Float16* cvct = (_Float16*)(ws + OFF_CVCT);
    float*    imp  = (float*)(ws + OFF_IMP);
    int*      lab  = (int*)(ws + OFF_LAB);
    int*      sel  = (int*)(ws + OFF_SEL);
    float*    bias = (float*)(ws + OFF_BIAS);
    int*      ksel = (int*)(ws + OFF_SCAL);
    int*      n32  = (int*)(ws + OFF_SCAL + 4);
    float*    qp   = (float*)(ws + OFF_QP);
    float*    part = (float*)(ws + OFF_PART);

    // 1) convert inputs to f16
    f32_to_f16_kernel<<<4096, 256, 0, stream>>>(X,  Xh,  (size_t)Mn * HDn);
    f32_to_f16_kernel<<<1024, 256, 0, stream>>>(Wq, Wqh, (size_t)HDn * HDn);
    f32_to_f16_kernel<<<1024, 256, 0, stream>>>(Wk, Wkh, (size_t)HDn * HDn);
    f32_to_f16_kernel<<<1024, 256, 0, stream>>>(Wv, Wvh, (size_t)HDn * HDn);
    f32_to_f16_kernel<<<1024, 256, 0, stream>>>(Wo, Woh, (size_t)HDn * HDn);

    // 2) projections (WMMA GEMMs)
    dim3 ggrid(Mn / 128, HDn / 128);
    gemm_wmma_kernel<<<ggrid, 256, 0, stream>>>(Xh, Wqh, nullptr, Qh, Mn, HDn, HDn);
    gemm_wmma_kernel<<<ggrid, 256, 0, stream>>>(Xh, Wkh, Kf,      Kh, Mn, HDn, HDn);
    gemm_wmma_kernel<<<ggrid, 256, 0, stream>>>(Xh, Wvh, Vf, nullptr, Mn, HDn, HDn);

    // 3) importance scores + labels
    bias_kernel<<<Bn, 64, 0, stream>>>(Qh, wimp, bimp, bias);
    importance_kernel<<<dim3(Bn, Sn / 128), 256, 0, stream>>>(Qh, Kh, pm, bias, imp, lab);

    // 4) median -> k_sel ; per-batch descending (val,idx) sort
    median_kernel<<<1, 1024, 0, stream>>>(imp, ksel, n32);
    topk_kernel<<<Bn, 1024, 0, stream>>>(imp, sel);

    // 5) per-level global min/max -> quant params ; quant-dequant + gather
    minmax1_kernel<<<256, 256, 0, stream>>>(Kf, Vf, lab, part);
    minmax2_kernel<<<1, 256, 0, stream>>>(part, qp);
    gather_quant_kernel<<<dim3(Sn, Bn), 256, 0, stream>>>(Kf, Vf, lab, sel, qp, ksel, ckc, cvct);

    // 6) flash attention over selected quantized KV
    attention_kernel<<<dim3(Sn / 128, Bn * Hn), 256, 0, stream>>>(Qh, ckc, cvct, ksel, n32, Ah);

    // 7) output projection
    gemm_wmma_kernel<<<ggrid, 256, 0, stream>>>(Ah, Woh, out, nullptr, Mn, HDn, HDn);
}
